// GraphUNet_53188874994306
// MI455X (gfx1250) — compile-verified, matrix operand order checked by reading, not run on hardware
//
#include <hip/hip_runtime.h>
#include <math.h>

// ---------------------------------------------------------------------------
// Types for CDNA5 WMMA (wave32, 16x16x32 bf16 -> f32)
// ---------------------------------------------------------------------------
typedef __attribute__((ext_vector_type(8)))  __bf16 v8bf;
typedef __attribute__((ext_vector_type(16))) __bf16 v16bf;
typedef __attribute__((ext_vector_type(8)))  float  v8f;

#define HEADS 2
#define NEG_SLOPE 0.2f

// ---------------------------------------------------------------------------
// Small device helpers
// ---------------------------------------------------------------------------
__device__ inline unsigned short f2bf(float f) {
  unsigned u = __float_as_uint(f);
  unsigned r = (u + 0x7FFFu + ((u >> 16) & 1u)) >> 16;  // RNE
  return (unsigned short)r;
}
__device__ inline float bf2f(unsigned short h) {
  return __uint_as_float(((unsigned)h) << 16);
}
// Monotone float -> uint key so atomicMax(uint) == float max.
__device__ inline unsigned fkey(float f) {
  unsigned u = __float_as_uint(f);
  return (u & 0x80000000u) ? ~u : (u | 0x80000000u);
}
__device__ inline float funkey(unsigned k) {
  unsigned u = (k & 0x80000000u) ? (k & 0x7FFFFFFFu) : ~k;
  return __uint_as_float(u);
}
#define FKEY_NEG_INF 0x007FFFFFu  // fkey(-inf)

// ---------------------------------------------------------------------------
// Pack fp32 (optionally concat of 2 sources, 2nd optionally batch-broadcast)
// into zero-padded bf16 row-major [M, Kp].
// ---------------------------------------------------------------------------
__global__ void k_pack(const float* __restrict__ s1, int c1,
                       const float* __restrict__ s2, int c2, int bcastRows,
                       unsigned short* __restrict__ out, int M, int Kp) {
  int i = blockIdx.x * blockDim.x + threadIdx.x;
  if (i >= M * Kp) return;
  int m = i / Kp, k = i - m * Kp;
  float v = 0.f;
  if (k < c1) {
    v = s1[(long)m * c1 + k];
  } else if (k < c1 + c2) {
    int r = bcastRows ? (m % bcastRows) : m;
    v = s2[(long)r * c2 + (k - c1)];
  }
  out[i] = f2bf(v);
}

// W[K,N] fp32 -> Wt[Np,Kp] bf16 (transposed, zero padded).
__global__ void k_wt(const float* __restrict__ W, int K, int N,
                     unsigned short* __restrict__ Wt, int Np, int Kp) {
  int i = blockIdx.x * blockDim.x + threadIdx.x;
  if (i >= Np * Kp) return;
  int n = i / Kp, k = i - n * Kp;
  float v = (n < N && k < K) ? W[(long)k * N + n] : 0.f;
  Wt[i] = f2bf(v);
}

// wa[k,h] = sum_d W[k, h*D+d] * a[h,d]   (rank-H fold of Wdst/We)
__global__ void k_wa(const float* __restrict__ W, int K, int N, int D,
                     const float* __restrict__ a, float* __restrict__ wa, int Kp) {
  int i = blockIdx.x * blockDim.x + threadIdx.x;
  if (i >= Kp * HEADS) return;
  int k = i / HEADS, h = i - k * HEADS;
  float s = 0.f;
  if (k < K)
    for (int d = 0; d < D; ++d) s += W[(long)k * N + h * D + d] * a[h * D + d];
  wa[i] = s;
}

// out[m,h] = sum_k bf16(A[m,k]) * wa[k,h]
__global__ void k_matvec(const unsigned short* __restrict__ A, int Kp,
                         const float* __restrict__ wa,
                         float* __restrict__ out, int M) {
  int i = blockIdx.x * blockDim.x + threadIdx.x;
  if (i >= M * HEADS) return;
  int m = i / HEADS, h = i - m * HEADS;
  const unsigned short* row = A + (long)m * Kp;
  float s = 0.f;
  for (int k = 0; k < Kp; ++k) s += bf2f(row[k]) * wa[k * HEADS + h];
  out[i] = s;
}

// a_s[m,h] = sum_d hs[m, h*D+d] * a[h,d]
__global__ void k_headdot(const float* __restrict__ hs, const float* __restrict__ a,
                          float* __restrict__ out, int M, int OUT, int D) {
  int i = blockIdx.x * blockDim.x + threadIdx.x;
  if (i >= M * HEADS) return;
  int m = i / HEADS, h = i - m * HEADS;
  const float* row = hs + (long)m * OUT + h * D;
  const float* av = a + h * D;
  float s = 0.f;
  for (int d = 0; d < D; ++d) s += row[d] * av[d];
  out[i] = s;
}

__global__ void k_fill(unsigned* __restrict__ p, unsigned v, long n) {
  long i = (long)blockIdx.x * blockDim.x + threadIdx.x;
  if (i < n) p[i] = v;
}

// Pass 1: logits + segment max (atomic on monotone key)
__global__ void k_edge_max(const int* __restrict__ ei, int E,
                           const float* __restrict__ a_s,
                           const float* __restrict__ a_d,
                           const float* __restrict__ a_e,
                           int Ns, int Nd, int adBatch,
                           float* __restrict__ lg, unsigned* __restrict__ mx, int B) {
  int i = blockIdx.x * blockDim.x + threadIdx.x;
  if (i >= B * E) return;
  int b = i / E, e = i - b * E;
  int s = ei[e], d = ei[E + e];
#pragma unroll
  for (int h = 0; h < HEADS; ++h) {
    float v = a_s[((long)b * Ns + s) * HEADS + h]
            + a_d[((long)(adBatch ? b * Nd : 0) + d) * HEADS + h]
            + a_e[(long)e * HEADS + h];
    v = v > 0.f ? v : NEG_SLOPE * v;
    lg[(long)i * HEADS + h] = v;
    atomicMax(&mx[((long)b * Nd + d) * HEADS + h], fkey(v));
  }
}

// Pass 2: exp(lg - max) + segment sum
__global__ void k_edge_exp(const int* __restrict__ ei, int E,
                           const float* __restrict__ lg,
                           const unsigned* __restrict__ mx,
                           float* __restrict__ ex, float* __restrict__ den,
                           int Nd, int B) {
  int i = blockIdx.x * blockDim.x + threadIdx.x;
  if (i >= B * E) return;
  int b = i / E, e = i - b * E;
  int d = ei[E + e];
#pragma unroll
  for (int h = 0; h < HEADS; ++h) {
    float m = funkey(mx[((long)b * Nd + d) * HEADS + h]);
    float t = __expf(lg[(long)i * HEADS + h] - m);
    ex[(long)i * HEADS + h] = t;
    atomicAdd(&den[((long)b * Nd + d) * HEADS + h], t);
  }
}

// Pass 3: acc[b,d,c] += hs[b,src,c] * alpha  (one thread per (b,e,c))
__global__ void k_scatter(const int* __restrict__ ei, int E,
                          const float* __restrict__ hs,
                          const float* __restrict__ ex,
                          const float* __restrict__ den,
                          float* __restrict__ acc,
                          int Ns, int Nd, int OUT, int D, int B) {
  long i = (long)blockIdx.x * blockDim.x + threadIdx.x;
  long tot = (long)B * E * OUT;
  if (i >= tot) return;
  int c = (int)(i % OUT);
  long t = i / OUT;
  int e = (int)(t % E);
  int b = (int)(t / E);
  int s = ei[e], d = ei[E + e];
  int h = c / D;
  float al = ex[((long)b * E + e) * HEADS + h] /
             (den[((long)b * Nd + d) * HEADS + h] + 1e-16f);
  atomicAdd(&acc[((long)b * Nd + d) * OUT + c],
            hs[((long)b * Ns + s) * OUT + c] * al);
}

// Epilogue: dst = acc + bias (+ res) (optionally gelu-tanh)
__global__ void k_epi(const float* __restrict__ acc, const float* __restrict__ bias,
                      const float* __restrict__ res, int resStride,
                      float* __restrict__ dst, long M, int OUT, int gelu) {
  long i = (long)blockIdx.x * blockDim.x + threadIdx.x;
  if (i >= M * OUT) return;
  int c = (int)(i % OUT);
  long m = i / OUT;
  float v = acc[i] + bias[c];
  if (res) v += res[m * (long)resStride + c];
  if (gelu) {
    float x = v;
    float u = 0.7978845608028654f * (x + 0.044715f * x * x * x);
    v = 0.5f * x * (1.f + tanhf(u));
  }
  dst[i] = v;
}

// ---------------------------------------------------------------------------
// WMMA GEMM: C[M,Nout] f32 = A[M,Kp]bf16 x Wt[Np,Kp]bf16^T
// 256 threads = 8 waves; each wave: 16 rows x NP cols. Wt staged in LDS.
// Requires M % 128 == 0 (true for every call site here).
// ---------------------------------------------------------------------------
template <int KP, int NP>
__global__ __launch_bounds__(256)
void k_gemm(const unsigned short* __restrict__ A,
            const unsigned short* __restrict__ Wt,
            float* __restrict__ C, long M, int Nout) {
  __shared__ unsigned short sw[NP * KP];
  int tid = threadIdx.x;
  for (int i = tid; i < NP * KP; i += 256) sw[i] = Wt[i];
  __syncthreads();

  int wave = tid >> 5, lane = tid & 31;
  long row0 = (long)blockIdx.x * 128 + wave * 16;
  int mrow = lane & 15;
  int kb = (lane < 16) ? 0 : 8;  // ISA 16-bit A/B fragment striping

  constexpr int NT = NP / 16;
  v8f acc[NT] = {};

  const unsigned short* arow = A + (row0 + mrow) * KP;
#pragma unroll
  for (int k0 = 0; k0 < KP; k0 += 32) {
    v8bf alo = *(const v8bf*)(const void*)(arow + k0 + kb);
    v8bf ahi = *(const v8bf*)(const void*)(arow + k0 + 16 + kb);
    v16bf av = __builtin_shufflevector(alo, ahi, 0, 1, 2, 3, 4, 5, 6, 7,
                                       8, 9, 10, 11, 12, 13, 14, 15);
#pragma unroll
    for (int t = 0; t < NT; ++t) {
      const unsigned short* bcol = sw + (t * 16 + (lane & 15)) * KP + k0 + kb;
      v8bf blo = *(const v8bf*)(const void*)(bcol);
      v8bf bhi = *(const v8bf*)(const void*)(bcol + 16);
      v16bf bv = __builtin_shufflevector(blo, bhi, 0, 1, 2, 3, 4, 5, 6, 7,
                                         8, 9, 10, 11, 12, 13, 14, 15);
      acc[t] = __builtin_amdgcn_wmma_f32_16x16x32_bf16(
          false, av, false, bv, (short)0, acc[t], false, false);
    }
  }

  // C/D layout: vgpr r, lanes 0-15 -> M=r, N=lane; lanes 16-31 -> M=8+r.
  int rlo = (lane >= 16) ? 8 : 0;
  int col0 = lane & 15;
#pragma unroll
  for (int t = 0; t < NT; ++t) {
    int col = t * 16 + col0;
    if (col < Nout) {
#pragma unroll
      for (int r = 0; r < 8; ++r) {
        long row = row0 + rlo + r;
        C[row * Nout + col] = acc[t][r];
      }
    }
  }
}

// ---------------------------------------------------------------------------
// Host-side orchestration
// ---------------------------------------------------------------------------
struct GatP { const float *Wsrc, *Wdst, *We, *a_src, *a_dst, *a_edge, *bias; };

struct Alloc { char* base; size_t off; size_t cap; };
static void* aget(Alloc& a, size_t bytes) {
  void* r = a.base + a.off;
  a.off += (bytes + 255) & ~(size_t)255;
  return r;
}

static inline dim3 g1(long n, int t) { return dim3((unsigned)((n + t - 1) / t)); }

static void launch_gemm(hipStream_t st, const unsigned short* A,
                        const unsigned short* Wt, float* C,
                        long M, int Kp, int Np, int Nout) {
  dim3 g((unsigned)(M / 128)), b(256);
  if (Kp == 128 && Np == 128)
    k_gemm<128, 128><<<g, b, 0, st>>>(A, Wt, C, M, Nout);
  else if (Kp == 160 && Np == 128)
    k_gemm<160, 128><<<g, b, 0, st>>>(A, Wt, C, M, Nout);
  else
    k_gemm<160, 112><<<g, b, 0, st>>>(A, Wt, C, M, Nout);
}

// One full GAT layer. Alloc passed by value: transients auto-free on return.
static void run_gat(hipStream_t st, Alloc al,
                    const float* s1, int c1, const float* s2, int c2, int bcastRows,
                    int B, int Ns, int Nd, int Kp,
                    const float* d1, int dc1, const float* d2, int dc2, int dKp,
                    const GatP& p, int OUT, int Np,
                    const int* ei, const float* ea, int E,
                    float* dst, const float* res, int resStride, int gelu) {
  const int H = HEADS, D = OUT / H;
  const long M = (long)B * Ns;
  const int Kact = c1 + c2;

  // hs = xs @ Wsrc  (the only full GEMM; WMMA)
  unsigned short* Abf = (unsigned short*)aget(al, (size_t)M * Kp * 2);
  k_pack<<<g1(M * Kp, 256), 256, 0, st>>>(s1, c1, s2, c2, bcastRows, Abf, (int)M, Kp);
  unsigned short* WsT = (unsigned short*)aget(al, (size_t)Np * Kp * 2);
  k_wt<<<g1((long)Np * Kp, 256), 256, 0, st>>>(p.Wsrc, Kact, OUT, WsT, Np, Kp);
  float* hs = (float*)aget(al, (size_t)M * OUT * 4);
  launch_gemm(st, Abf, WsT, hs, M, Kp, Np, OUT);

  // a_s
  float* a_s = (float*)aget(al, (size_t)M * H * 4);
  k_headdot<<<g1(M * H, 256), 256, 0, st>>>(hs, p.a_src, a_s, (int)M, OUT, D);

  // a_d via rank-H folded Wdst
  float* a_d;
  int adBatch;
  if (d1) {  // mapper: xd = [ctx, ll], batch-shared
    adBatch = 0;
    float* wd_a = (float*)aget(al, (size_t)dKp * H * 4);
    k_wa<<<g1((long)dKp * H, 64), 64, 0, st>>>(p.Wdst, dc1 + dc2, OUT, D, p.a_dst, wd_a, dKp);
    unsigned short* Adbf = (unsigned short*)aget(al, (size_t)Nd * dKp * 2);
    k_pack<<<g1((long)Nd * dKp, 256), 256, 0, st>>>(d1, dc1, d2, dc2, 0, Adbf, Nd, dKp);
    a_d = (float*)aget(al, (size_t)Nd * H * 4);
    k_matvec<<<g1((long)Nd * H, 256), 256, 0, st>>>(Adbf, dKp, wd_a, a_d, Nd);
  } else {  // encoder: xd == xs
    adBatch = 1;
    float* wd_a = (float*)aget(al, (size_t)Kp * H * 4);
    k_wa<<<g1((long)Kp * H, 64), 64, 0, st>>>(p.Wdst, Kact, OUT, D, p.a_dst, wd_a, Kp);
    a_d = (float*)aget(al, (size_t)M * H * 4);
    k_matvec<<<g1(M * H, 256), 256, 0, st>>>(Abf, Kp, wd_a, a_d, (int)M);
  }

  // a_e via rank-H folded We (K=3)
  float* we_a = (float*)aget(al, (size_t)32 * H * 4);
  k_wa<<<g1(32 * H, 64), 64, 0, st>>>(p.We, 3, OUT, D, p.a_edge, we_a, 32);
  unsigned short* Aebf = (unsigned short*)aget(al, (size_t)E * 32 * 2);
  k_pack<<<g1((long)E * 32, 256), 256, 0, st>>>(ea, 3, nullptr, 0, 0, Aebf, E, 32);
  float* a_e = (float*)aget(al, (size_t)E * H * 4);
  k_matvec<<<g1((long)E * H, 256), 256, 0, st>>>(Aebf, 32, we_a, a_e, E);

  // segment softmax over edges
  unsigned* mx = (unsigned*)aget(al, (size_t)B * Nd * H * 4);
  k_fill<<<g1((long)B * Nd * H, 256), 256, 0, st>>>(mx, FKEY_NEG_INF, (long)B * Nd * H);
  float* lg = (float*)aget(al, (size_t)B * E * H * 4);
  k_edge_max<<<g1((long)B * E, 256), 256, 0, st>>>(ei, E, a_s, a_d, a_e, Ns, Nd,
                                                   adBatch, lg, mx, B);
  float* den = (float*)aget(al, (size_t)B * Nd * H * 4);
  k_fill<<<g1((long)B * Nd * H, 256), 256, 0, st>>>((unsigned*)den, 0u, (long)B * Nd * H);
  float* ex = (float*)aget(al, (size_t)B * E * H * 4);
  k_edge_exp<<<g1((long)B * E, 256), 256, 0, st>>>(ei, E, lg, mx, ex, den, Nd, B);

  // weighted scatter
  float* acc = (float*)aget(al, (size_t)B * Nd * OUT * 4);
  k_fill<<<g1((long)B * Nd * OUT, 256), 256, 0, st>>>((unsigned*)acc, 0u, (long)B * Nd * OUT);
  k_scatter<<<g1((long)B * E * OUT, 256), 256, 0, st>>>(ei, E, hs, ex, den, acc,
                                                        Ns, Nd, OUT, D, B);

  // bias (+ residual / gelu)
  k_epi<<<g1((long)B * Nd * OUT, 256), 256, 0, st>>>(acc, p.bias, res, resStride,
                                                     dst, (long)B * Nd, OUT, gelu);
}

extern "C" void kernel_launch(void* const* d_in, const int* in_sizes, int n_in,
                              void* d_out, int out_size, void* d_ws, size_t ws_size,
                              hipStream_t stream) {
  (void)in_sizes; (void)n_in; (void)out_size;
  const int N160 = 108160, N80 = 28480, N48 = 10944, N32 = 5248;
  const int B = 2;

  auto F = [&](int i) { return (const float*)d_in[i]; };
  auto I = [&](int i) { return (const int*)d_in[i]; };
  auto gp = [&](int b) { return GatP{F(b), F(b+1), F(b+2), F(b+3), F(b+4), F(b+5), F(b+6)}; };

  const float* X = F(0);                                 // (2,108160,114)
  GatP m160_80 = gp(1);  const float* ctx80  = F(8);
  GatP m80_48  = gp(9);  const float* ctx48  = F(16);
  GatP m48_32  = gp(17); const float* ctx32  = F(24);
  GatP m32_48  = gp(25); const float* ctx48b = F(32);
  GatP m48_80  = gp(33); const float* ctx80b = F(40);
  GatP m80_160 = gp(41); const float* ctx160 = F(48);
  GatP e80L0 = gp(49), e80L1 = gp(56);
  GatP e48L0 = gp(63), e48L1 = gp(70);
  GatP e32L0 = gp(77), e32L1 = gp(84);
  const int *ei160_80 = I(91), *ei80_80 = I(93), *ei80_48 = I(95), *ei48_48 = I(97),
            *ei48_32 = I(99), *ei32_32 = I(101), *ei32_48 = I(103), *ei48_80 = I(105),
            *ei80_160 = I(107);
  const float *ea160_80 = F(92), *ea80_80 = F(94), *ea80_48 = F(96), *ea48_48 = F(98),
              *ea48_32 = F(100), *ea32_32 = F(102), *ea32_48 = F(104), *ea48_80 = F(106),
              *ea80_160 = F(108);
  const float *ll160 = F(109), *ll80 = F(110), *ll48 = F(111), *ll32 = F(112);
  const int E160_80 = 340000, E80_80 = 256320, E80_48 = 98496, E48_48 = 98496,
            E48_32 = 47232, E32_32 = 47232, E32_48 = 98496, E48_80 = 256320,
            E80_160 = 324480;

  Alloc al{(char*)d_ws, 0, ws_size};
  // Persistent activations (fp32, stride 128)
  float* X80  = (float*)aget(al, (size_t)B * N80 * 128 * 4);
  float* X80P = (float*)aget(al, (size_t)B * N80 * 128 * 4);
  float* X48A = (float*)aget(al, (size_t)B * N48 * 128 * 4);
  float* X48P = (float*)aget(al, (size_t)B * N48 * 128 * 4);
  float* X32A = (float*)aget(al, (size_t)B * N32 * 128 * 4);
  float* X32P = (float*)aget(al, (size_t)B * N32 * 128 * 4);
  float* X48U = (float*)aget(al, (size_t)B * N48 * 128 * 4);
  float* X80U = (float*)aget(al, (size_t)B * N80 * 128 * 4);

  // ---- x80 = mapper m160_80(concat(x, ll160)) ----
  run_gat(stream, al, X, 114, ll160, 4, N160, B, N160, N80, 128,
          ctx80, 1, ll80, 4, 32, m160_80, 128, 128,
          ei160_80, ea160_80, E160_80, X80, nullptr, 0, 0);

  // ---- x80p = enc80(x80) + x80 ----
  {
    Alloc t = al;
    float* h1 = (float*)aget(t, (size_t)B * N80 * 128 * 4);
    run_gat(stream, t, X80, 128, nullptr, 0, 0, B, N80, N80, 128,
            nullptr, 0, nullptr, 0, 0, e80L0, 128, 128,
            ei80_80, ea80_80, E80_80, h1, nullptr, 0, 1);
    run_gat(stream, t, h1, 128, nullptr, 0, 0, B, N80, N80, 128,
            nullptr, 0, nullptr, 0, 0, e80L1, 128, 128,
            ei80_80, ea80_80, E80_80, X80P, X80, 128, 0);
  }

  // ---- x48 = mapper m80_48(concat(x80p, ll80)) ----
  run_gat(stream, al, X80P, 128, ll80, 4, N80, B, N80, N48, 160,
          ctx48, 1, ll48, 4, 32, m80_48, 128, 128,
          ei80_48, ea80_48, E80_48, X48A, nullptr, 0, 0);

  // ---- x48p = enc48(x48) + x48 ----
  {
    Alloc t = al;
    float* h1 = (float*)aget(t, (size_t)B * N48 * 128 * 4);
    run_gat(stream, t, X48A, 128, nullptr, 0, 0, B, N48, N48, 128,
            nullptr, 0, nullptr, 0, 0, e48L0, 128, 128,
            ei48_48, ea48_48, E48_48, h1, nullptr, 0, 1);
    run_gat(stream, t, h1, 128, nullptr, 0, 0, B, N48, N48, 128,
            nullptr, 0, nullptr, 0, 0, e48L1, 128, 128,
            ei48_48, ea48_48, E48_48, X48P, X48A, 128, 0);
  }

  // ---- x32 = mapper m48_32(concat(x48p, ll48)) ----
  run_gat(stream, al, X48P, 128, ll48, 4, N48, B, N48, N32, 160,
          ctx32, 1, ll32, 4, 32, m48_32, 128, 128,
          ei48_32, ea48_32, E48_32, X32A, nullptr, 0, 0);

  // ---- x32p = enc32(x32) + x32 ----
  {
    Alloc t = al;
    float* h1 = (float*)aget(t, (size_t)B * N32 * 128 * 4);
    run_gat(stream, t, X32A, 128, nullptr, 0, 0, B, N32, N32, 128,
            nullptr, 0, nullptr, 0, 0, e32L0, 128, 128,
            ei32_32, ea32_32, E32_32, h1, nullptr, 0, 1);
    run_gat(stream, t, h1, 128, nullptr, 0, 0, B, N32, N32, 128,
            nullptr, 0, nullptr, 0, 0, e32L1, 128, 128,
            ei32_32, ea32_32, E32_32, X32P, X32A, 128, 0);
  }

  // ---- x48u = mapper m32_48(concat(x32p, ll32)) + x48p ----
  run_gat(stream, al, X32P, 128, ll32, 4, N32, B, N32, N48, 160,
          ctx48b, 1, ll48, 4, 32, m32_48, 128, 128,
          ei32_48, ea32_48, E32_48, X48U, X48P, 128, 0);

  // ---- x80u = mapper m48_80(concat(x48u, ll48)) + x80p ----
  run_gat(stream, al, X48U, 128, ll48, 4, N48, B, N48, N80, 160,
          ctx80b, 1, ll80, 4, 32, m48_80, 128, 128,
          ei48_80, ea48_80, E48_80, X80U, X80P, 128, 0);

  // ---- out = mapper m80_160(concat(x80u, ll80)) + x[..., :98] ----
  run_gat(stream, al, X80U, 128, ll80, 4, N80, B, N80, N160, 160,
          ctx160, 1, ll160, 4, 32, m80_160, 98, 112,
          ei80_160, ea80_160, E80_160, (float*)d_out, X, 114, 0);
}
